// ResidualBlock_62732292325818
// MI455X (gfx1250) — matrix operand order repaired and checked by reading, not compiled, not run on hardware
//
#include <hip/hip_runtime.h>
#include <math.h>

// ---------------- CDNA5 WMMA types ----------------
typedef __attribute__((ext_vector_type(16))) __bf16 v16bf;
typedef __attribute__((ext_vector_type(8)))  float  v8f;

#define TT 2048
#define DD 2048
#define WWID 2048
#define HH 8192

// ---------------- helpers ----------------
__device__ __forceinline__ unsigned short f2bf(float f) {
  unsigned u = __float_as_uint(f);
  u += 0x7FFFu + ((u >> 16) & 1u);          // round-to-nearest-even
  return (unsigned short)(u >> 16);
}
__device__ __forceinline__ float bf2f(unsigned short h) {
  return __uint_as_float(((unsigned)h) << 16);
}
__device__ __forceinline__ float sigmoid_f(float x) { return 1.0f / (1.0f + expf(-x)); }
__device__ __forceinline__ float gelu_f(float x) {
  return 0.5f * x * (1.0f + erff(x * 0.7071067811865475f));
}

union FragB { v16bf v; uint4 q[2]; };

// ---------------- rmsnorm + bf16 hi/lo split ----------------
__global__ __launch_bounds__(256)
void rmsnorm_split_kernel(const float* __restrict__ x, const float* __restrict__ g,
                          unsigned short* __restrict__ hi, unsigned short* __restrict__ lo,
                          int D)
{
  __shared__ float red[8];
  __shared__ float inv_s;
  const int row = blockIdx.x;
  const float* xr = x + (size_t)row * D;
  const int tid = threadIdx.x;

  float s = 0.f;
  for (int i = tid; i < D; i += 256) { float v = xr[i]; s += v * v; }
#pragma unroll
  for (int off = 16; off > 0; off >>= 1) s += __shfl_xor(s, off, 32);
  if ((tid & 31) == 0) red[tid >> 5] = s;
  __syncthreads();
  if (tid == 0) {
    float t = 0.f;
#pragma unroll
    for (int i = 0; i < 8; ++i) t += red[i];
    inv_s = 45.254833995939045f / fmaxf(sqrtf(t), 1e-12f);  // sqrt(D)=sqrt(2048)
  }
  __syncthreads();
  const float inv = inv_s;
  for (int i = tid; i < D; i += 256) {
    float y = xr[i] * inv * g[i];
    unsigned short h = f2bf(y);
    size_t idx = (size_t)row * D + i;
    hi[idx] = h;
    lo[idx] = f2bf(y - bf2f(h));
  }
}

// ---------------- depthwise conv1d (k=4, pad=2, slice[:T]) + split ----------------
__global__ __launch_bounds__(256)
void conv_split_kernel(const float* __restrict__ xr, const float* __restrict__ cw,
                       const float* __restrict__ cb,
                       float* __restrict__ xcf,
                       unsigned short* __restrict__ hi, unsigned short* __restrict__ lo,
                       int T, int W)
{
  int idx = blockIdx.x * 256 + threadIdx.x;
  if (idx >= T * W) return;
  int t = idx / W;
  int c = idx - t * W;
  float s = cb[c];
#pragma unroll
  for (int j = 0; j < 4; ++j) {
    int ts = t - 2 + j;
    if (ts >= 0 && ts < T) s += cw[c * 4 + j] * xr[(size_t)ts * W + c];
  }
  xcf[idx] = s;
  unsigned short h = f2bf(s);
  hi[idx] = h;
  lo[idx] = f2bf(s - bf2f(h));
}

// ---------------- WMMA GEMM: C[M,N] = A[M,K] * Wt[N,K]^T (+epilogue) ----------------
// MODE 0: outF = acc + b1
// MODE 1: outF = acc + b1 + aux1   (residual add)
// MODE 2: dual (Wa,Wx) + RG-LRU epilogue -> split(outHi/outLo)
//         rt=sig(acc1+b1); it=sig(acc2+b2); la=-log1p(exp(-Lam));
//         a=exp(la*rt/8); ht=a*h0 + sqrt(1-a^2)*(it*xc=aux2); comb=gelu(aux1)*ht
// MODE 3: dual (mlp1,mlp2): comb = gelu(sig(acc1+b1))*(acc2+b2) -> split
template<int MODE>
__global__ __launch_bounds__(256)
void gemm_kernel(const unsigned short* __restrict__ Ahi,
                 const unsigned short* __restrict__ Alo,
                 const float* __restrict__ Wm1, const float* __restrict__ bias1,
                 const float* __restrict__ Wm2, const float* __restrict__ bias2,
                 float* __restrict__ outF,
                 unsigned short* __restrict__ outHi, unsigned short* __restrict__ outLo,
                 const float* __restrict__ aux1, const float* __restrict__ aux2,
                 const float* __restrict__ h0v, const float* __restrict__ Lam,
                 int M, int N, int K)
{
  constexpr bool DUAL = (MODE == 2 || MODE == 3);
  constexpr int  BN = DUAL ? 64 : 128;   // block tile N
  constexpr int  WN = DUAL ? 32 : 64;    // wave tile N
  constexpr int  NF = WN / 16;           // N fragments per wave
  constexpr int  KT = 32;                // K step
  constexpr int  KP = 40;                // padded LDS row (bf16 elems)

  __shared__ unsigned short sA[2][128 * KP];                 // hi/lo A tile
  __shared__ unsigned short sB1[2][BN * KP];                 // hi/lo W1 tile
  __shared__ unsigned short sB2[DUAL ? 2 : 1][DUAL ? (BN * KP) : 8];

  const int tid  = threadIdx.x;
  const int lane = tid & 31;
  const int wid  = tid >> 5;
  const int waveM = wid >> 1;            // 0..3
  const int waveN = wid & 1;             // 0..1
  const int bm0 = blockIdx.y * 128;
  const int bn0 = blockIdx.x * BN;

  v8f acc1[2][NF];
  v8f acc2[DUAL ? 2 : 1][NF];
#pragma unroll
  for (int i = 0; i < 2; ++i)
#pragma unroll
    for (int j = 0; j < NF; ++j) {
      acc1[i][j] = (v8f){0.f, 0.f, 0.f, 0.f, 0.f, 0.f, 0.f, 0.f};
      if constexpr (DUAL) acc2[i][j] = (v8f){0.f, 0.f, 0.f, 0.f, 0.f, 0.f, 0.f, 0.f};
    }

  for (int k0 = 0; k0 < K; k0 += KT) {
    __syncthreads();
    // ---- stage A (already bf16 hi/lo planes): 128 x 32 ----
#pragma unroll
    for (int itr = 0; itr < 2; ++itr) {
      int slot = tid + itr * 256;        // 0..511
      int r = slot >> 2;                 // 0..127
      int c = (slot & 3) * 8;            // 0,8,16,24
      size_t g = (size_t)(bm0 + r) * K + k0 + c;
      *(uint4*)&sA[0][r * KP + c] = *(const uint4*)(Ahi + g);
      *(uint4*)&sA[1][r * KP + c] = *(const uint4*)(Alo + g);
    }
    // ---- stage W (f32 -> bf16 hi/lo split on the fly): BN x 32 ----
    constexpr int BITERS = (BN * 8) / 256;
#pragma unroll
    for (int itr = 0; itr < BITERS; ++itr) {
      int slot = tid + itr * 256;
      int r = slot >> 3;                 // 0..BN-1
      int c = (slot & 7) * 4;            // 0..28
      size_t g = (size_t)(bn0 + r) * K + k0 + c;
      {
        float4 w = *(const float4*)(Wm1 + g);
        unsigned short h0s = f2bf(w.x), h1s = f2bf(w.y), h2s = f2bf(w.z), h3s = f2bf(w.w);
        unsigned short l0 = f2bf(w.x - bf2f(h0s)), l1 = f2bf(w.y - bf2f(h1s));
        unsigned short l2 = f2bf(w.z - bf2f(h2s)), l3 = f2bf(w.w - bf2f(h3s));
        uint2 ph, pl;
        ph.x = (unsigned)h0s | ((unsigned)h1s << 16);
        ph.y = (unsigned)h2s | ((unsigned)h3s << 16);
        pl.x = (unsigned)l0 | ((unsigned)l1 << 16);
        pl.y = (unsigned)l2 | ((unsigned)l3 << 16);
        *(uint2*)&sB1[0][r * KP + c] = ph;
        *(uint2*)&sB1[1][r * KP + c] = pl;
      }
      if constexpr (DUAL) {
        float4 w = *(const float4*)(Wm2 + g);
        unsigned short h0s = f2bf(w.x), h1s = f2bf(w.y), h2s = f2bf(w.z), h3s = f2bf(w.w);
        unsigned short l0 = f2bf(w.x - bf2f(h0s)), l1 = f2bf(w.y - bf2f(h1s));
        unsigned short l2 = f2bf(w.z - bf2f(h2s)), l3 = f2bf(w.w - bf2f(h3s));
        uint2 ph, pl;
        ph.x = (unsigned)h0s | ((unsigned)h1s << 16);
        ph.y = (unsigned)h2s | ((unsigned)h3s << 16);
        pl.x = (unsigned)l0 | ((unsigned)l1 << 16);
        pl.y = (unsigned)l2 | ((unsigned)l3 << 16);
        *(uint2*)&sB2[0][r * KP + c] = ph;
        *(uint2*)&sB2[1][r * KP + c] = pl;
      }
    }
    if (k0 + KT < K) {                    // global_prefetch_b8 of next weight chunk
      __builtin_prefetch((const void*)(Wm1 + (size_t)(bn0 + (tid & (BN - 1))) * K + k0 + KT), 0, 1);
      if constexpr (DUAL)
        __builtin_prefetch((const void*)(Wm2 + (size_t)(bn0 + (tid & (BN - 1))) * K + k0 + KT), 0, 1);
    }
    __syncthreads();

    // ---- fragments (CDNA5 wave32 layouts) ----
    const int mrow = lane & 15;
    const int kgrp = lane >> 4;
    FragB ah[2], al[2];
#pragma unroll
    for (int mi = 0; mi < 2; ++mi) {
      int r = waveM * 32 + mi * 16 + mrow;
      // A 16x32 bf16: lane holds contiguous 32B of A row at K=(L>>4)*16
      ah[mi].q[0] = *(const uint4*)&sA[0][r * KP + kgrp * 16];
      ah[mi].q[1] = *(const uint4*)&sA[0][r * KP + kgrp * 16 + 8];
      al[mi].q[0] = *(const uint4*)&sA[1][r * KP + kgrp * 16];
      al[mi].q[1] = *(const uint4*)&sA[1][r * KP + kgrp * 16 + 8];
    }
#pragma unroll
    for (int nj = 0; nj < NF; ++nj) {
      int rB = waveN * WN + nj * 16 + mrow;  // row of W (== column n of B)
      FragB bh, bl;
      // B 32x16: lane holds contiguous 32B of W row at K=(L>>4)*16
      bh.q[0] = *(const uint4*)&sB1[0][rB * KP + kgrp * 16];
      bh.q[1] = *(const uint4*)&sB1[0][rB * KP + kgrp * 16 + 8];
      bl.q[0] = *(const uint4*)&sB1[1][rB * KP + kgrp * 16];
      bl.q[1] = *(const uint4*)&sB1[1][rB * KP + kgrp * 16 + 8];
#pragma unroll
      for (int mi = 0; mi < 2; ++mi) {
        acc1[mi][nj] = __builtin_amdgcn_wmma_f32_16x16x32_bf16(
            false, ah[mi].v, false, bh.v, (short)0, acc1[mi][nj], false, false);
        acc1[mi][nj] = __builtin_amdgcn_wmma_f32_16x16x32_bf16(
            false, ah[mi].v, false, bl.v, (short)0, acc1[mi][nj], false, false);
        acc1[mi][nj] = __builtin_amdgcn_wmma_f32_16x16x32_bf16(
            false, al[mi].v, false, bh.v, (short)0, acc1[mi][nj], false, false);
      }
      if constexpr (DUAL) {
        FragB ch, cl;
        ch.q[0] = *(const uint4*)&sB2[0][rB * KP + kgrp * 16];
        ch.q[1] = *(const uint4*)&sB2[0][rB * KP + kgrp * 16 + 8];
        cl.q[0] = *(const uint4*)&sB2[1][rB * KP + kgrp * 16];
        cl.q[1] = *(const uint4*)&sB2[1][rB * KP + kgrp * 16 + 8];
#pragma unroll
        for (int mi = 0; mi < 2; ++mi) {
          acc2[mi][nj] = __builtin_amdgcn_wmma_f32_16x16x32_bf16(
              false, ah[mi].v, false, ch.v, (short)0, acc2[mi][nj], false, false);
          acc2[mi][nj] = __builtin_amdgcn_wmma_f32_16x16x32_bf16(
              false, ah[mi].v, false, cl.v, (short)0, acc2[mi][nj], false, false);
          acc2[mi][nj] = __builtin_amdgcn_wmma_f32_16x16x32_bf16(
              false, al[mi].v, false, ch.v, (short)0, acc2[mi][nj], false, false);
        }
      }
    }
  }

  // ---- epilogue: D layout lane L, vgpr v -> (m = (L>>4)*8+v, n = L&15) ----
#pragma unroll
  for (int mi = 0; mi < 2; ++mi)
#pragma unroll
    for (int nj = 0; nj < NF; ++nj)
#pragma unroll
      for (int v = 0; v < 8; ++v) {
        int row = bm0 + waveM * 32 + mi * 16 + (lane >> 4) * 8 + v;
        int col = bn0 + waveN * WN + nj * 16 + (lane & 15);
        size_t idx = (size_t)row * N + col;
        float z1 = acc1[mi][nj][v] + bias1[col];
        if constexpr (MODE == 0) {
          outF[idx] = z1;
        } else if constexpr (MODE == 1) {
          outF[idx] = z1 + aux1[idx];
        } else if constexpr (MODE == 2) {
          float z2  = acc2[mi][nj][v] + bias2[col];
          float rt  = sigmoid_f(z1);
          float itv = sigmoid_f(z2);
          float la  = -log1pf(expf(-Lam[col]));
          float a   = expf(la * rt * 0.125f);
          float xc  = aux2[idx];
          float ht  = a * h0v[col] + sqrtf(fmaxf(1.0f - a * a, 0.0f)) * (itv * xc);
          float cb  = gelu_f(aux1[idx]) * ht;
          unsigned short hh = f2bf(cb);
          outHi[idx] = hh;
          outLo[idx] = f2bf(cb - bf2f(hh));
        } else {
          float z2 = acc2[mi][nj][v] + bias2[col];
          float cb = gelu_f(sigmoid_f(z1)) * z2;
          unsigned short hh = f2bf(cb);
          outHi[idx] = hh;
          outLo[idx] = f2bf(cb - bf2f(hh));
        }
      }
}

// ---------------- host orchestration ----------------
extern "C" void kernel_launch(void* const* d_in, const int* in_sizes, int n_in,
                              void* d_out, int out_size, void* d_ws, size_t ws_size,
                              hipStream_t stream)
{
  (void)in_sizes; (void)n_in; (void)out_size; (void)ws_size;
  const float* x   = (const float*)d_in[0];
  const float* h0  = (const float*)d_in[1];
  const float* g1  = (const float*)d_in[2];
  const float* g2  = (const float*)d_in[3];
  const float* Wrl = (const float*)d_in[4];
  const float* brl = (const float*)d_in[5];
  const float* cw  = (const float*)d_in[6];
  const float* cb  = (const float*)d_in[7];
  const float* Wa  = (const float*)d_in[8];
  const float* ba  = (const float*)d_in[9];
  const float* Wx  = (const float*)d_in[10];
  const float* bx  = (const float*)d_in[11];
  const float* Lam = (const float*)d_in[12];
  const float* W2r = (const float*)d_in[13];
  const float* b2r = (const float*)d_in[14];
  const float* Wm1 = (const float*)d_in[15];
  const float* bm1 = (const float*)d_in[16];
  const float* Wm2 = (const float*)d_in[17];
  const float* bm2 = (const float*)d_in[18];
  const float* Wm3 = (const float*)d_in[19];
  const float* bm3 = (const float*)d_in[20];
  float* out = (float*)d_out;

  char* p = (char*)d_ws;
  auto alloc = [&](size_t bytes) -> void* {
    void* r = (void*)p;
    p += (bytes + 255) & ~(size_t)255;
    return r;
  };
  unsigned short* xn_hi  = (unsigned short*)alloc((size_t)TT * DD * 2);
  unsigned short* xn_lo  = (unsigned short*)alloc((size_t)TT * DD * 2);
  float*          xr     = (float*)alloc((size_t)TT * WWID * 4);
  float*          xcf    = (float*)alloc((size_t)TT * WWID * 4);
  unsigned short* xc_hi  = (unsigned short*)alloc((size_t)TT * WWID * 2);
  unsigned short* xc_lo  = (unsigned short*)alloc((size_t)TT * WWID * 2);
  unsigned short* cmb_hi = (unsigned short*)alloc((size_t)TT * WWID * 2);
  unsigned short* cmb_lo = (unsigned short*)alloc((size_t)TT * WWID * 2);
  float*          resid  = (float*)alloc((size_t)TT * DD * 4);
  unsigned short* x2_hi  = (unsigned short*)alloc((size_t)TT * DD * 2);
  unsigned short* x2_lo  = (unsigned short*)alloc((size_t)TT * DD * 2);
  unsigned short* mc_hi  = (unsigned short*)alloc((size_t)TT * HH * 2);
  unsigned short* mc_lo  = (unsigned short*)alloc((size_t)TT * HH * 2);

  // 1) xn = rmsnorm(x, g1), split
  rmsnorm_split_kernel<<<TT, 256, 0, stream>>>(x, g1, xn_hi, xn_lo, DD);
  // 2) xr = xn @ rec_lin_w^T + b
  gemm_kernel<0><<<dim3(WWID / 128, TT / 128), 256, 0, stream>>>(
      xn_hi, xn_lo, Wrl, brl, nullptr, nullptr, xr, nullptr, nullptr,
      nullptr, nullptr, nullptr, nullptr, TT, WWID, DD);
  // 3) depthwise conv -> xc (f32 + split)
  conv_split_kernel<<<(TT * WWID) / 256, 256, 0, stream>>>(xr, cw, cb, xcf, xc_hi, xc_lo, TT, WWID);
  // 4) dual GEMM (Wa, Wx) + RG-LRU epilogue -> combined (split)
  gemm_kernel<2><<<dim3(WWID / 64, TT / 128), 256, 0, stream>>>(
      xc_hi, xc_lo, Wa, ba, Wx, bx, nullptr, cmb_hi, cmb_lo,
      xr, xcf, h0, Lam, TT, WWID, WWID);
  // 5) xrec = combined @ rec_lin2^T + b + x -> residual
  gemm_kernel<1><<<dim3(DD / 128, TT / 128), 256, 0, stream>>>(
      cmb_hi, cmb_lo, W2r, b2r, nullptr, nullptr, resid, nullptr, nullptr,
      x, nullptr, nullptr, nullptr, TT, DD, WWID);
  // 6) xn2 = rmsnorm(residual, g2), split
  rmsnorm_split_kernel<<<TT, 256, 0, stream>>>(resid, g2, x2_hi, x2_lo, DD);
  // 7) dual GEMM (mlp1, mlp2) + gated-MLP combine -> comb (split)
  gemm_kernel<3><<<dim3(HH / 64, TT / 128), 256, 0, stream>>>(
      x2_hi, x2_lo, Wm1, bm1, Wm2, bm2, nullptr, mc_hi, mc_lo,
      nullptr, nullptr, nullptr, nullptr, TT, HH, DD);
  // 8) out = comb @ mlp3^T + b + residual
  gemm_kernel<1><<<dim3(DD / 128, TT / 128), 256, 0, stream>>>(
      mc_hi, mc_lo, Wm3, bm3, nullptr, nullptr, out, nullptr, nullptr,
      resid, nullptr, nullptr, nullptr, TT, DD, HH);
}
